// AdaptiveVectorModifier_70952859730356
// MI455X (gfx1250) — compile-verified, matrix operand order checked by reading, not run on hardware
//
#include <hip/hip_runtime.h>
#include <stdint.h>

typedef __bf16 bf16;
typedef __attribute__((ext_vector_type(16))) __bf16 v16bf;
typedef __attribute__((ext_vector_type(8)))  __bf16 v8bf;
typedef __attribute__((ext_vector_type(4)))  __bf16 v4bf;
typedef __attribute__((ext_vector_type(8)))  float  v8f;
typedef __attribute__((ext_vector_type(4)))  int    v4i;

#define N_ROWS 8192
#define VDIM   2048
#define MDIM   128

// ---------- WMMA helpers ----------
union V16U { v16bf v; v8bf h[2]; };

__device__ __forceinline__ v8f wmma_bf16(v16bf a, v16bf b, v8f c) {
  return __builtin_amdgcn_wmma_f32_16x16x32_bf16(false, a, false, b, (short)0, c,
                                                 false, false);
}

// Load one 16(free-dim) x 32(K) bf16 operand fragment, row-major with given
// element stride.  Lane l: row = l%16, kbase = k0 + 8*(l/16); elements
// [kbase..kbase+7] and [kbase+16..kbase+23] (CDNA5 16-bit A layout).
__device__ __forceinline__ v16bf load_frag(const bf16* base, int stride, int k0) {
  int l = threadIdx.x & 31;
  const bf16* p = base + (int64_t)(l & 15) * stride + k0 + ((l >> 4) << 3);
  V16U u;
  u.h[0] = *(const v8bf*)(p);
  u.h[1] = *(const v8bf*)(p + 16);
  return u.v;
}

// ---------- async LDS copy (CDNA5) ----------
#if defined(__HIP_DEVICE_COMPILE__) && __has_builtin(__builtin_amdgcn_global_load_async_to_lds_b128)
#define HAVE_ASYNC 1
typedef __attribute__((address_space(1))) v4i* g4i_t;   // global <4 x i32>*
typedef __attribute__((address_space(3))) v4i* l4i_t;   // LDS <4 x i32>*
__device__ __forceinline__ g4i_t to_glob(const void* p) {
  return (g4i_t)(uintptr_t)p;
}
__device__ __forceinline__ l4i_t to_lds(const void* p) {
  return (l4i_t)(uint32_t)(uintptr_t)p;  // low 32 bits of flat LDS addr = LDS offset
}
#endif

__device__ __forceinline__ void wait_async0() {
#if defined(__HIP_DEVICE_COMPILE__) && __has_builtin(__builtin_amdgcn_s_wait_asynccnt)
  __builtin_amdgcn_s_wait_asynccnt(0);
#elif defined(__HIP_DEVICE_COMPILE__)
  asm volatile("s_wait_asynccnt 0x0" ::: "memory");
#endif
}
__device__ __forceinline__ void wait_async32() {
#if defined(__HIP_DEVICE_COMPILE__) && __has_builtin(__builtin_amdgcn_s_wait_asynccnt)
  __builtin_amdgcn_s_wait_asynccnt(32);
#elif defined(__HIP_DEVICE_COMPILE__)
  asm volatile("s_wait_asynccnt 0x20" ::: "memory");
#endif
}

// ---------- k0: convert weights to bf16 scratch copies ----------
__global__ void k0_convert(const float* __restrict__ Wmap, const float* __restrict__ W1,
                           const float* __restrict__ W2, bf16* __restrict__ Wmap_bf,
                           bf16* __restrict__ WmapT_bf, bf16* __restrict__ W1_bf,
                           bf16* __restrict__ W2_bf) {
  int64_t stride = (int64_t)gridDim.x * blockDim.x;
  int64_t tid = (int64_t)blockIdx.x * blockDim.x + threadIdx.x;
  for (int64_t t = tid; t < (int64_t)MDIM * VDIM; t += stride) {
    float v = Wmap[t];
    Wmap_bf[t] = (bf16)v;
    int64_t r = t >> 11, c = t & (VDIM - 1);
    WmapT_bf[c * MDIM + r] = (bf16)v;     // transposed copy for the back-map GEMM
  }
  for (int64_t t = tid; t < (int64_t)512 * MDIM; t += stride) W1_bf[t] = (bf16)W1[t];
  for (int64_t t = tid; t < (int64_t)16384 * 512; t += stride) W2_bf[t] = (bf16)W2[t];
}

// ---------- k1: feats = x @ Wmap^T  (per-WG 16 rows, x tile staged bf16 in LDS) ----------
#define XSTRIDE 2056  // 2048 + 8 bf16 pad (bank-conflict free, 16B aligned rows)
__global__ void k1_feats(const float* __restrict__ x, const bf16* __restrict__ Wmap_bf,
                         bf16* __restrict__ feats_bf) {
  extern __shared__ char smem[];
  bf16* xs = (bf16*)smem;                 // [16][XSTRIDE]
  int n0 = blockIdx.x * 16;
  for (int q = threadIdx.x; q < 16 * 512; q += 256) {
    int r = q >> 9, c4 = (q & 511) << 2;
    float4 xv = *(const float4*)(x + (int64_t)(n0 + r) * VDIM + c4);
    v4bf o = {(bf16)xv.x, (bf16)xv.y, (bf16)xv.z, (bf16)xv.w};
    *(v4bf*)(xs + r * XSTRIDE + c4) = o;
  }
  __syncthreads();
  int w = threadIdx.x >> 5, l = threadIdx.x & 31, half = l >> 4;
  int c0 = w * 16;                        // 8 waves cover all 128 feature cols
  v8f acc = {0.f, 0.f, 0.f, 0.f, 0.f, 0.f, 0.f, 0.f};
#pragma unroll 4
  for (int ks = 0; ks < 64; ++ks) {
    v16bf a = load_frag(xs, XSTRIDE, 32 * ks);
    v16bf b = load_frag(Wmap_bf + (int64_t)c0 * VDIM, VDIM, 32 * ks);
    acc = wmma_bf16(a, b, acc);
  }
  int c = c0 + (l & 15);
#pragma unroll
  for (int v = 0; v < 8; ++v)
    feats_bf[(int64_t)(n0 + half * 8 + v) * MDIM + c] = (bf16)acc[v];
}

// ---------- k2: h = silu(feats @ W1^T + b1) ----------
__global__ void k2_h(const bf16* __restrict__ feats_bf, const bf16* __restrict__ W1_bf,
                     const float* __restrict__ b1, bf16* __restrict__ h_bf) {
  int n0 = blockIdx.x * 16;
  int w = threadIdx.x >> 5, l = threadIdx.x & 31, half = l >> 4;
  v16bf a[4];
#pragma unroll
  for (int ks = 0; ks < 4; ++ks)
    a[ks] = load_frag(feats_bf + (int64_t)n0 * MDIM, MDIM, 32 * ks);
  for (int q = 0; q < 4; ++q) {
    int c0 = (w * 4 + q) * 16;            // 8 waves x 4 tiles = 512 cols
    v8f acc = {0.f, 0.f, 0.f, 0.f, 0.f, 0.f, 0.f, 0.f};
#pragma unroll
    for (int ks = 0; ks < 4; ++ks)
      acc = wmma_bf16(a[ks], load_frag(W1_bf + (int64_t)c0 * MDIM, MDIM, 32 * ks), acc);
    int c = c0 + (l & 15);
    float bias = b1[c];
#pragma unroll
    for (int v = 0; v < 8; ++v) {
      float z = acc[v] + bias;
      float s = z / (1.f + __expf(-z));   // silu
      h_bf[(int64_t)(n0 + half * 8 + v) * 512 + c] = (bf16)s;
    }
  }
}

// ---------- k3: fused  A = h@W2^T (+b2)  and  feats2 = A . feats ----------
// WG: 128 rows (8 waves x 16) x 32 i-values.  D(16p x 16n) = W2tile x h^T.
#define W2STRIDE 520                       // 512 + 8 bf16 pad
#define W2BUF (128 * W2STRIDE)
#define FSTRIDE 136                        // 128 + 8 bf16 pad
#define K3_LDS (2 * W2BUF * 2 + 128 * FSTRIDE * 2)

__device__ __forceinline__ void stage_w2(const bf16* __restrict__ W2_bf, bf16* dst, int i) {
  int t = threadIdx.x;
  int row = t >> 1, hsel = t & 1;          // 2 threads per W2 row, 512 B each
  const bf16* src = W2_bf + (((int64_t)i * 128 + row) << 9) + hsel * 256;
  bf16* d = dst + row * W2STRIDE + hsel * 256;
#if defined(HAVE_ASYNC)
#pragma unroll
  for (int c = 0; c < 32; ++c)
    __builtin_amdgcn_global_load_async_to_lds_b128(to_glob(src + c * 8),
                                                   to_lds(d + c * 8), 0, 0);
#else
#pragma unroll
  for (int c = 0; c < 32; ++c)
    *(v8bf*)(d + c * 8) = *(const v8bf*)(src + c * 8);
#endif
}

__global__ void __launch_bounds__(256) k3_main(const bf16* __restrict__ h_bf,
                                               const bf16* __restrict__ W2_bf,
                                               const float* __restrict__ b2,
                                               const bf16* __restrict__ feats_bf,
                                               bf16* __restrict__ feats2_bf) {
  extern __shared__ char smem[];
  bf16* w2s = (bf16*)smem;                        // double buffer, 2 x 128 x W2STRIDE
  bf16* fsh = (bf16*)(smem + 2 * W2BUF * 2);      // feats tile, 128 x FSTRIDE
  int rb = blockIdx.x & 63, ic = blockIdx.x >> 6;
  int w = threadIdx.x >> 5, l = threadIdx.x & 31, half = l >> 4, nl = l & 15;
  int n0wg = rb * 128;
  int n0 = n0wg + w * 16;

  // h fragments for this wave's 16 rows live in registers for the whole kernel
  v16bf hreg[16];
#pragma unroll
  for (int ks = 0; ks < 16; ++ks)
    hreg[ks] = load_frag(h_bf + (int64_t)n0 * 512, 512, 32 * ks);

  // feats tile (bf16) -> LDS, cooperative
  for (int q = threadIdx.x; q < 2048; q += 256) {
    int r = q >> 4, cc = (q & 15) << 3;
    v8bf fv = *(const v8bf*)(feats_bf + (int64_t)(n0wg + r) * MDIM + cc);
    *(v8bf*)(fsh + r * FSTRIDE + cc) = fv;
  }

  int i0 = ic * 32;
  stage_w2(W2_bf, w2s, i0);                       // prologue fill of buffer 0
  for (int ii = 0; ii < 32; ++ii) {
    bf16* cur = w2s + (ii & 1) * W2BUF;
    bf16* nxt = w2s + ((ii + 1) & 1) * W2BUF;
    bool pf = (ii + 1) < 32;
    if (pf) { stage_w2(W2_bf, nxt, i0 + ii + 1); wait_async32(); }
    else    { wait_async0(); }
    __syncthreads();

    int i = i0 + ii;
    float partial = 0.f;
    for (int jb = 0; jb < 8; ++jb) {
      v8f acc = {0.f, 0.f, 0.f, 0.f, 0.f, 0.f, 0.f, 0.f};
      const bf16* arow = cur + (jb * 16) * W2STRIDE;
#pragma unroll
      for (int ks = 0; ks < 16; ++ks)
        acc = wmma_bf16(load_frag(arow, W2STRIDE, 32 * ks), hreg[ks], acc);
      // A values for pairs p = 8*half + v (j = 16*jb + p), this lane's n = nl
      const float* b2p = b2 + i * 128 + jb * 16 + half * 8;
      float4 bA = *(const float4*)b2p;
      float4 bB = *(const float4*)(b2p + 4);
      float bb[8] = {bA.x, bA.y, bA.z, bA.w, bB.x, bB.y, bB.z, bB.w};
      const bf16* fp = fsh + (w * 16 + nl) * FSTRIDE + jb * 16 + half * 8;
      v8bf fv = *(const v8bf*)fp;
#pragma unroll
      for (int v = 0; v < 8; ++v)
        partial += (acc[v] + bb[v]) * (float)fv[v];
    }
    partial += __shfl_xor(partial, 16, 32);       // combine the two lane halves
    if (half == 0)
      feats2_bf[(int64_t)(n0 + nl) * MDIM + i] = (bf16)partial;
    __syncthreads();
  }
}

// ---------- k4: out = x + feats2 @ Wmap ----------
__global__ void k4_out(const float* __restrict__ x, const bf16* __restrict__ feats2_bf,
                       const bf16* __restrict__ WmapT_bf, float* __restrict__ out) {
  int n0 = blockIdx.x * 16;
  int w = threadIdx.x >> 5, l = threadIdx.x & 31, half = l >> 4;
  v16bf a[4];
#pragma unroll
  for (int ks = 0; ks < 4; ++ks)
    a[ks] = load_frag(feats2_bf + (int64_t)n0 * MDIM, MDIM, 32 * ks);
  for (int t = 0; t < 16; ++t) {
    int c0 = (w * 16 + t) * 16;                   // 8 waves x 16 tiles = 2048 cols
    v8f acc = {0.f, 0.f, 0.f, 0.f, 0.f, 0.f, 0.f, 0.f};
#pragma unroll
    for (int ks = 0; ks < 4; ++ks)
      acc = wmma_bf16(a[ks], load_frag(WmapT_bf + (int64_t)c0 * MDIM, MDIM, 32 * ks), acc);
    int c = c0 + (l & 15);
#pragma unroll
    for (int v = 0; v < 8; ++v) {
      int64_t idx = (int64_t)(n0 + half * 8 + v) * VDIM + c;
      out[idx] = x[idx] + acc[v];
    }
  }
}

// ---------- host launcher ----------
extern "C" void kernel_launch(void* const* d_in, const int* in_sizes, int n_in,
                              void* d_out, int out_size, void* d_ws, size_t ws_size,
                              hipStream_t stream) {
  const float* x    = (const float*)d_in[0];
  const float* Wmap = (const float*)d_in[1];
  const float* W1   = (const float*)d_in[2];
  const float* b1   = (const float*)d_in[3];
  const float* W2   = (const float*)d_in[4];
  const float* b2   = (const float*)d_in[5];
  float* out = (float*)d_out;

  char* ws = (char*)d_ws;
  bf16* W2_bf     = (bf16*)(ws);               // 16,777,216 B
  bf16* Wmap_bf   = (bf16*)(ws + 16777216);    //    524,288 B
  bf16* WmapT_bf  = (bf16*)(ws + 17301504);    //    524,288 B
  bf16* W1_bf     = (bf16*)(ws + 17825792);    //    131,072 B
  bf16* feats_bf  = (bf16*)(ws + 17956864);    //  2,097,152 B
  bf16* h_bf      = (bf16*)(ws + 20054016);    //  8,388,608 B
  bf16* feats2_bf = (bf16*)(ws + 28442624);    //  2,097,152 B

  k0_convert<<<1024, 256, 0, stream>>>(Wmap, W1, W2, Wmap_bf, WmapT_bf, W1_bf, W2_bf);
  k1_feats<<<N_ROWS / 16, 256, 16 * XSTRIDE * 2, stream>>>(x, Wmap_bf, feats_bf);
  k2_h<<<N_ROWS / 16, 256, 0, stream>>>(feats_bf, W1_bf, b1, h_bf);
  k3_main<<<(N_ROWS / 128) * 4, 256, K3_LDS, stream>>>(h_bf, W2_bf, b2, feats_bf, feats2_bf);
  k4_out<<<N_ROWS / 16, 256, 0, stream>>>(x, feats2_bf, WmapT_bf, out);
}
